// CustomLSTMCell_10264971837806
// MI455X (gfx1250) — matrix operand order repaired and checked by reading, not compiled, not run on hardware
//
#include <hip/hip_runtime.h>
#include <math.h>

// ---------------------------------------------------------------------------
// CDNA5 (gfx1250) LSTM: persistent wave32 WMMA kernel, explicitly
// register-resident weights (256 VGPRs of B fragments), x-half GEMM
// software-pipelined into the grid-barrier wait.
// ---------------------------------------------------------------------------

typedef __attribute__((ext_vector_type(16))) __bf16 v16bf;
typedef __attribute__((ext_vector_type(8)))  float  v8f;

#define HDIM    512
#define T_STEPS 1024
#define BATCH   64
#define NBLK    64      // 32 hidden tiles x 2 batch halves

// round-to-nearest-even fp32 -> bf16 bit pattern
__device__ __forceinline__ unsigned short f2bf(float x) {
  unsigned u = __builtin_bit_cast(unsigned, x);
  unsigned r = (u + 0x7FFFu + ((u >> 16) & 1u)) >> 16;
  return (unsigned short)r;
}

union Frag { v16bf v; uint4 u[2]; };

__device__ __forceinline__ float sigmoidf_(float x) {
  return 1.f / (1.f + __expf(-x));
}

// ---------------------------------------------------------------------------
// Prep kernel 1: x_seq fp32 -> bf16 (vectorized 4/thread)
// ---------------------------------------------------------------------------
__global__ void cvt_x(const float* __restrict__ x, unsigned short* __restrict__ xb, int n4) {
  int i = blockIdx.x * blockDim.x + threadIdx.x;
  if (i < n4) {
    float4 v = ((const float4*)x)[i];
    ushort4 o;
    o.x = f2bf(v.x); o.y = f2bf(v.y); o.z = f2bf(v.z); o.w = f2bf(v.w);
    ((ushort4*)xb)[i] = o;
  }
}

// ---------------------------------------------------------------------------
// Prep kernel 2: pack weights into Wb[gate*512+j][k] (K-major, K = [h(512)|x(512)])
// as bf16, and fuse biases: bias[g] = b_h[g] + b_x[g]
// ---------------------------------------------------------------------------
struct WPtrs {
  const float* wh[4];
  const float* wx[4];
  const float* bh[4];
  const float* bx[4];
};

__global__ void pack_w(WPtrs p, unsigned short* __restrict__ Wb, float* __restrict__ bias) {
  int idx = blockIdx.x * blockDim.x + threadIdx.x;   // 0 .. 2048*1024-1
  if (idx >= 2048 * 1024) return;
  int g    = idx >> 10;     // gate row 0..2047
  int k    = idx & 1023;    // K 0..1023
  int gate = g >> 9;        // 0:f 1:i 2:g 3:o
  int j    = g & 511;
  float v = (k < 512) ? p.wh[gate][j * 512 + k]
                      : p.wx[gate][j * 512 + (k - 512)];
  Wb[idx] = f2bf(v);
  if (k == 0) bias[g] = p.bh[gate][j] + p.bx[gate][j];
}

// ---------------------------------------------------------------------------
// Prep kernel 3: zero h double-buffer (2 x 64x512 bf16) + barrier counters
// ---------------------------------------------------------------------------
__global__ void init_state(unsigned short* __restrict__ h, unsigned* __restrict__ sync) {
  int i = blockIdx.x * blockDim.x + threadIdx.x;
  if (i < 2 * BATCH * HDIM) h[i] = 0;
  if (i == 0) { sync[0] = 0u; sync[1] = 0u; }
}

// x-half of the gate GEMM: K = 512..1023, independent of h -> runs inside the
// barrier wait. B fragments come from the register-resident array.
__device__ __forceinline__ v8f x_gemm(const Frag (&Bx)[16],
                                      const unsigned short* __restrict__ Ax) {
  v8f a = {0.f, 0.f, 0.f, 0.f, 0.f, 0.f, 0.f, 0.f};
#pragma unroll
  for (int ks = 0; ks < 16; ++ks) {
    Frag A;
    const unsigned short* ap = Ax + ks * 32;
    A.u[0] = *(const uint4*)(ap);
    A.u[1] = *(const uint4*)(ap + 16);
    a = __builtin_amdgcn_wmma_f32_16x16x32_bf16(
        false, A.v, false, Bx[ks].v, (short)0, a, false, false);
  }
  return a;
}

// ---------------------------------------------------------------------------
// Persistent recurrent kernel.
//  grid = 64 blocks x 256 threads (8 wave32)
//  block (jblk, mhalf): hidden units j in [jblk*16, jblk*16+16), batch rows
//  b in [mhalf*32, mhalf*32+32).
//  Wave w: gate q = w&3 (N-tile), batch tile mt = (w>>2) + 2*mhalf.
// ---------------------------------------------------------------------------
__global__ __launch_bounds__(256) void lstm_seq(
    const unsigned short* __restrict__ xb,    // [64][1024][512] bf16
    const unsigned short* __restrict__ Wb,    // [2048][1024]    bf16
    const float*          __restrict__ bias,  // [2048]
    unsigned short*       __restrict__ h0,    // [64][512] bf16 (ping)
    unsigned short*       __restrict__ h1,    // [64][512] bf16 (pong)
    unsigned*             __restrict__ sync,  // {count, generation}
    float*                __restrict__ out)   // [64][512] fp32
{
  __shared__ float gbuf[4][32][16];   // gate accums: [gate][b_local][j_local]
  __shared__ float sb[4][16];         // fused biases for this block's 16 units

  const int tid   = threadIdx.x;
  const int blk   = blockIdx.x;
  const int jblk  = blk >> 1;              // 0..31
  const int mhalf = blk & 1;               // 0..1
  const int wv    = tid >> 5;              // wave 0..7
  const int lane  = tid & 31;
  const int q     = wv & 3;                // gate / N-tile
  const int mt    = (wv >> 2) + 2 * mhalf; // batch tile 0..3
  const int lj    = lane & 15;
  const int t16   = lane >> 4;

  if (tid < 64) {
    int g = tid >> 4, jj = tid & 15;
    sb[g][jj] = bias[g * HDIM + jblk * 16 + jj];
  }

  // B fragment base: lane holds column (gate row) g = q*512 + jblk*16 + lj,
  // chunks K = k0 + t16*8 + [0,8) and k0 + 16 + t16*8 + [0,8) (Wb is K-major).
  const unsigned short* Brow =
      Wb + (size_t)(q * HDIM + jblk * 16 + lj) * 1024 + t16 * 8;

  // Preload ALL weight fragments into registers: 32 frags x 8 VGPRs = 256 VGPRs.
  // Survives the per-step cache invalidate; removes B loads from the loop.
  Frag Bh[16], Bx[16];
#pragma unroll
  for (int ks = 0; ks < 16; ++ks) {
    const unsigned short* bp = Brow + ks * 32;
    Bh[ks].u[0] = *(const uint4*)(bp);
    Bh[ks].u[1] = *(const uint4*)(bp + 16);
    const unsigned short* bq = Brow + 512 + ks * 32;
    Bx[ks].u[0] = *(const uint4*)(bq);
    Bx[ks].u[1] = *(const uint4*)(bq + 16);
  }

  // A fragment: lane holds batch row bg = mt*16 + lj;
  // chunks K = k0 + t16*8 + [0,8) and k0 + 16 + t16*8 + [0,8).
  const int bg = mt * 16 + lj;
  const unsigned short* Axb = xb + (size_t)bg * (T_STEPS * HDIM) + t16 * 8;

  float creg[2] = {0.f, 0.f};
  float hreg[2] = {0.f, 0.f};

  __syncthreads();

  // Prologue: x-half accumulator for t = 0
  v8f accx = x_gemm(Bx, Axb);

  for (int t = 0; t < T_STEPS; ++t) {
    const unsigned short* hr = (t & 1) ? h1 : h0;
    unsigned short*       hw = (t & 1) ? h0 : h1;

    // ---- h-half (critical path: depends on previous step) ----
    v8f acc = accx;
    const unsigned short* Ah = hr + bg * HDIM + t16 * 8;
#pragma unroll
    for (int ks = 0; ks < 16; ++ks) {
      Frag A;
      const unsigned short* ap = Ah + ks * 32;
      A.u[0] = *(const uint4*)(ap);
      A.u[1] = *(const uint4*)(ap + 16);
      acc = __builtin_amdgcn_wmma_f32_16x16x32_bf16(
          false, A.v, false, Bh[ks].v, (short)0, acc, false, false);
    }

    // C/D layout: VGPR r -> M = t16*8 + r (within tile), N = lj
    {
      const int blrow = (wv >> 2) * 16 + t16 * 8;   // batch row local to block
#pragma unroll
      for (int r = 0; r < 8; ++r)
        gbuf[q][blrow + r][lj] = acc[r];
    }
    __syncthreads();

    // ---- cell update: 512 cells (32 b x 16 j), 2/thread; c in registers ----
#pragma unroll
    for (int i = 0; i < 2; ++i) {
      int cell = tid + i * 256;
      int bl = cell >> 4, jj = cell & 15;
      float fg = sigmoidf_(gbuf[0][bl][jj] + sb[0][jj]);
      float ig = sigmoidf_(gbuf[1][bl][jj] + sb[1][jj]);
      float gg = tanhf    (gbuf[2][bl][jj] + sb[2][jj]);
      float og = sigmoidf_(gbuf[3][bl][jj] + sb[3][jj]);
      float c = fg * creg[i] + ig * gg;
      creg[i] = c;
      float h = og * tanhf(c);
      hreg[i] = h;
      int b = mhalf * 32 + bl;
      int j = jblk * 16 + jj;
      hw[b * HDIM + j] = f2bf(h);
    }

    // ---- release: ALL threads flush h stores to device scope ----
    __builtin_amdgcn_fence(__ATOMIC_RELEASE, "agent");
    __syncthreads();

    // ---- grid barrier, split signal/wait with x-GEMM overlapped ----
    unsigned g0 = 0u;
    bool last = false;
    if (tid == 0) {
      g0 = __hip_atomic_load(&sync[1], __ATOMIC_RELAXED, __HIP_MEMORY_SCOPE_AGENT);
      unsigned a = __hip_atomic_fetch_add(&sync[0], 1u, __ATOMIC_ACQ_REL, __HIP_MEMORY_SCOPE_AGENT);
      if (a == (unsigned)(NBLK - 1)) {
        __hip_atomic_store(&sync[0], 0u, __ATOMIC_RELAXED, __HIP_MEMORY_SCOPE_AGENT);
        __hip_atomic_fetch_add(&sync[1], 1u, __ATOMIC_RELEASE, __HIP_MEMORY_SCOPE_AGENT);
        last = true;
      }
    }

    // Overlap: next step's x-half GEMM (independent of h) hides barrier latency.
    if (t + 1 < T_STEPS)
      accx = x_gemm(Bx, Axb + (t + 1) * HDIM);

    if (tid == 0 && !last) {
      while (__hip_atomic_load(&sync[1], __ATOMIC_ACQUIRE, __HIP_MEMORY_SCOPE_AGENT) == g0)
        __builtin_amdgcn_s_sleep(1);
    }
    __syncthreads();
    // ---- acquire: every wave invalidates its caches before reading new h ----
    __builtin_amdgcn_fence(__ATOMIC_ACQUIRE, "agent");
  }

  // final hidden state in fp32
#pragma unroll
  for (int i = 0; i < 2; ++i) {
    int cell = tid + i * 256;
    int bl = cell >> 4, jj = cell & 15;
    int b = mhalf * 32 + bl;
    int j = jblk * 16 + jj;
    out[b * HDIM + j] = hreg[i];
  }
}

// ---------------------------------------------------------------------------
// Launch
// ---------------------------------------------------------------------------
extern "C" void kernel_launch(void* const* d_in, const int* in_sizes, int n_in,
                              void* d_out, int out_size, void* d_ws, size_t ws_size,
                              hipStream_t stream) {
  (void)in_sizes; (void)n_in; (void)out_size; (void)ws_size;

  const float* x = (const float*)d_in[0];

  // workspace layout (bytes)
  char* ws = (char*)d_ws;
  unsigned short* h0   = (unsigned short*)(ws + 0);        //  64 KB
  unsigned short* h1   = (unsigned short*)(ws + 65536);    //  64 KB
  unsigned*       sync = (unsigned*)      (ws + 131072);   // 256 B
  float*          bias = (float*)         (ws + 131328);   //   8 KB
  unsigned short* Wb   = (unsigned short*)(ws + 139520);   //   4 MB
  unsigned short* xb   = (unsigned short*)(ws + 4333824);  //  64 MB

  WPtrs p;
  // dict order: x_seq, then (W_hf,b_hf),(W_xf,b_xf),(W_hi,b_hi),(W_xi,b_xi),
  //             (W_hg,b_hg),(W_xg,b_xg),(W_ho,b_ho),(W_xo,b_xo)
  p.wh[0] = (const float*)d_in[1];  p.bh[0] = (const float*)d_in[2];
  p.wx[0] = (const float*)d_in[3];  p.bx[0] = (const float*)d_in[4];
  p.wh[1] = (const float*)d_in[5];  p.bh[1] = (const float*)d_in[6];
  p.wx[1] = (const float*)d_in[7];  p.bx[1] = (const float*)d_in[8];
  p.wh[2] = (const float*)d_in[9];  p.bh[2] = (const float*)d_in[10];
  p.wx[2] = (const float*)d_in[11]; p.bx[2] = (const float*)d_in[12];
  p.wh[3] = (const float*)d_in[13]; p.bh[3] = (const float*)d_in[14];
  p.wx[3] = (const float*)d_in[15]; p.bx[3] = (const float*)d_in[16];

  const int n4 = (BATCH * T_STEPS * HDIM) / 4;  // 8,388,608 float4 groups
  cvt_x<<<n4 / 256, 256, 0, stream>>>(x, xb, n4);
  pack_w<<<(2048 * 1024) / 256, 256, 0, stream>>>(p, Wb, bias);
  init_state<<<256, 256, 0, stream>>>(h0, sync);
  lstm_seq<<<NBLK, 256, 0, stream>>>(xb, Wb, bias, h0, h1, sync, (float*)d_out);
}
